// SwarmByteRingModel_51608327028848
// MI455X (gfx1250) — compile-verified, matrix operand order checked by reading, not run on hardware
//
#include <hip/hip_runtime.h>
#include <math.h>

// Problem constants (match reference)
#define B_   16
#define T_   512
#define P_   256
#define E_   128
#define NB_  4
#define KOFF 4
#define AS_  136   // padded staging row stride in halves (16B-aligned rows, bank rotation)

typedef __attribute__((ext_vector_type(16))) _Float16 v16h;
typedef __attribute__((ext_vector_type(8)))  _Float16 v8h;
typedef __attribute__((ext_vector_type(4)))  _Float16 v4h;
typedef __attribute__((ext_vector_type(8)))  float    v8f;
typedef __attribute__((ext_vector_type(4)))  float    v4f;

union HFrag { v16h v; v8h h2[2]; _Float16 h[16]; };

__device__ __forceinline__ float fast_rcp(float x) { return __builtin_amdgcn_rcpf(x); }

// Branch-free tanh: clamp + v_exp_f32 + v_rcp_f32 (no div_scale/div_fmas ladder)
__device__ __forceinline__ float fast_tanh(float x) {
    const float xc  = fminf(fmaxf(x, -10.0f), 10.0f);
    const float e2x = __expf(2.0f * xc);
    return (e2x - 1.0f) * fast_rcp(e2x + 1.0f);
}
__device__ __forceinline__ float fast_sigmoid(float x) {
    return fast_rcp(1.0f + __expf(-x));
}

__global__ void zero_ring_kernel(float* __restrict__ p, int n) {
    int i = blockIdx.x * blockDim.x + threadIdx.x;
    if (i < n) p[i] = 0.0f;
}

// Single persistent workgroup: 256 threads = 8 wave32 waves.
// Per block-step: 8 waves x 4 WMMAs for s@W_p (16x128x128), plus wave 0 does
// 4 WMMAs for the fused output projection s2@[W_out|0] (16x128x16).
__global__ __launch_bounds__(256)
void swarm_ring_kernel(const float* __restrict__ x,
                       const float* __restrict__ W_in,  const float* __restrict__ b_in,
                       const float* __restrict__ W_out, const float* __restrict__ b_out,
                       const float* __restrict__ W_p,   const float* __restrict__ b_p,
                       const float* __restrict__ ptr_dest,
                       const float* __restrict__ jump_W, const float* __restrict__ jump_b,
                       const float* __restrict__ ctx_strength,
                       const float* __restrict__ phase_bias,
                       const float* __restrict__ pointer_init,
                       float* __restrict__ out, float* __restrict__ ring)
{
    __shared__ __align__(16) _Float16 sA [B_ * AS_];   // s1 staged as f16 (WMMA A)
    __shared__ __align__(16) _Float16 sA2[B_ * AS_];   // s2 staged as f16 (proj WMMA A)
    __shared__ __align__(16) _Float16 sWoutF[4 * 32 * 16]; // [W_out|0] B-fragment image
    __shared__ __align__(16) float sInp[B_ * E_];      // x_t @ W_in + b_in
    __shared__ __align__(16) float sHid[NB_ * B_ * E_];// per-block recurrent state (== s2)
    __shared__ __align__(16) float sJW[NB_ * E_];
    __shared__ __align__(16) float sPhase[NB_ * E_];   // pre-scaled by 0.1
    __shared__ float sW[B_ * 9];                       // softmax neighbor weights
    __shared__ int   sIdx[B_ * 9];                     // neighbor ring indices
    __shared__ float sPtr[NB_ * B_];                   // pointers
    __shared__ float sAcc[B_ * 8];                     // output accumulator over blocks
    __shared__ float sBp[E_], sBin[E_];
    __shared__ float sBout[8];
    __shared__ float sJB[NB_], sCS[NB_];

    const int tid  = threadIdx.x;
    const int lane = tid & 31;
    const int wave = tid >> 5;

    // ---- stage constants into LDS, init recurrent state ----
    for (int i = tid; i < E_;       i += 256) { sBin[i] = b_in[i]; sBp[i] = b_p[i]; }
    for (int i = tid; i < NB_ * E_; i += 256) { sJW[i] = jump_W[i];
                                                sPhase[i] = 0.1f * phase_bias[i]; }
    if (tid < 8)   sBout[tid] = b_out[tid];
    if (tid < NB_) { sJB[tid] = jump_b[tid]; sCS[tid] = fast_sigmoid(ctx_strength[tid]); }
    for (int i = tid; i < NB_ * B_;      i += 256) sPtr[i] = pointer_init[i];
    for (int i = tid; i < NB_ * B_ * E_; i += 256) sHid[i] = 0.0f;

    // ---- pre-pack [W_out | 0] (128x16) B-fragments into LDS (once) ----
    if (tid < 128) {
        const int kc = tid >> 5, ln = tid & 31;
        const int n = ln & 15, kb2 = (ln >> 4) * 8;
        const int k0 = kc * 32 + kb2;
        _Float16* dst = &sWoutF[(kc * 32 + ln) * 16];
        for (int j = 0; j < 8; ++j) {
            dst[j]     = (n < 8) ? (_Float16)W_out[(k0 + j)      * 8 + n] : (_Float16)0.0f;
            dst[8 + j] = (n < 8) ? (_Float16)W_out[(k0 + 16 + j) * 8 + n] : (_Float16)0.0f;
        }
    }

    // ---- pre-pack W_p B-fragments (f16) into VGPRs, resident for all 2048 steps ----
    const int n0   = wave * 16;
    const int ncol = n0 + (lane & 15);
    const int kb   = (lane >> 4) * 8;
    HFrag bfr[4];
    #pragma unroll
    for (int kc = 0; kc < 4; ++kc) {
        const int k0 = kc * 32 + kb;
        #pragma unroll
        for (int j = 0; j < 8; ++j) {
            bfr[kc].h[j]     = (_Float16)W_p[(k0 + j)      * E_ + ncol];
            bfr[kc].h[8 + j] = (_Float16)W_p[(k0 + 16 + j) * E_ + ncol];
        }
    }
    __syncthreads();

    for (int t = 0; t < T_; ++t) {
        // ---- inp = x_t @ W_in + b_in (K=8, VALU, v4f vectorized) ----
        for (int p4 = tid; p4 < B_ * E_ / 4; p4 += 256) {
            const int b = p4 >> 5, e4 = (p4 & 31) * 4;
            const float* xr = x + ((size_t)b * T_ + t) * 8;
            v4f acc = *(const v4f*)&sBin[e4];
            #pragma unroll
            for (int k = 0; k < 8; ++k) acc += xr[k] * *(const v4f*)&W_in[k * E_ + e4];
            *(v4f*)&sInp[b * E_ + e4] = acc;
        }
        if (tid < B_ * 8) sAcc[tid] = 0.0f;
        __syncthreads();

        for (int blk = 0; blk < NB_; ++blk) {
            // ---- A: pointer neighborhood softmax (1 thread / batch) + ring prefetch ----
            if (tid < B_) {
                const int b = tid;
                const float ptr = sPtr[blk * B_ + b];
                int base = (int)floorf(ptr);
                base = min(max(base, 0), P_ - 1);
                float ex[9]; float mx = -1e30f;
                #pragma unroll
                for (int j = 0; j < 9; ++j) {
                    const int idx = (base + j - KOFF + P_) & (P_ - 1);
                    sIdx[b * 9 + j] = idx;
                    // wrap (idx - ptr + 128) into [0,256) with selects (range-limited)
                    float d = (float)idx - ptr + 128.0f;
                    d = (d >= 256.0f) ? d - 256.0f : d;
                    d = (d < 0.0f)    ? d + 256.0f : d;
                    d -= 128.0f;
                    const float e2 = -d * d * (1.0f / 8.0f);
                    ex[j] = e2;
                    mx = fmaxf(mx, e2);
                    const float* row = &ring[((size_t)b * P_ + idx) * E_];
                    __builtin_prefetch(row, 0, 1);       // global_prefetch_b8: warm caches
                    __builtin_prefetch(row + 64, 0, 1);  // second half of 512B row
                }
                float s = 0.0f;
                #pragma unroll
                for (int j = 0; j < 9; ++j) { ex[j] = __expf(ex[j] - mx); s += ex[j]; }
                const float inv = fast_rcp(s);
                #pragma unroll
                for (int j = 0; j < 9; ++j) sW[b * 9 + j] = ex[j] * inv;
            }
            __syncthreads();

            // ---- B: v4f ring gather, s1 = tanh(inp + cs*ctx + phase01 + hid) ----
            for (int p4 = tid; p4 < B_ * E_ / 4; p4 += 256) {
                const int b = p4 >> 5, e4 = (p4 & 31) * 4;
                v4f ctx = {};
                #pragma unroll
                for (int j = 0; j < 9; ++j)
                    ctx += sW[b * 9 + j] *
                           *(const v4f*)&ring[((size_t)b * P_ + sIdx[b * 9 + j]) * E_ + e4];
                const v4f comb = *(const v4f*)&sInp[b * E_ + e4] + sCS[blk] * ctx
                               + *(const v4f*)&sPhase[blk * E_ + e4]
                               + *(const v4f*)&sHid[(blk * B_ + b) * E_ + e4];
                v4h h4;
                #pragma unroll
                for (int c = 0; c < 4; ++c) h4[c] = (_Float16)fast_tanh(comb[c]);
                *(v4h*)&sA[b * AS_ + e4] = h4;
            }
            __syncthreads();

            // ---- C: s2 = tanh(s1 @ W_p + b_p) via 4 chained WMMAs per wave ----
            {
                v8f c = {};
                const int m = lane & 15;
                #pragma unroll
                for (int kc = 0; kc < 4; ++kc) {
                    HFrag a;
                    const int koff = kc * 32 + kb;
                    a.h2[0] = *(const v8h*)&sA[m * AS_ + koff];
                    a.h2[1] = *(const v8h*)&sA[m * AS_ + koff + 16];
                    c = __builtin_amdgcn_wmma_f32_16x16x32_f16(
                            false, a.v, false, bfr[kc].v, (short)0, c, false, false);
                }
                // C/D layout: VGPR r -> M = r + 8*(lane>=16); N = lane&15
                const int   mbase = (lane >> 4) * 8;
                const float bpn   = sBp[ncol];
                #pragma unroll
                for (int r = 0; r < 8; ++r) {
                    const float v = fast_tanh(c[r] + bpn);
                    sHid[(blk * B_ + (mbase + r)) * E_ + ncol] = v;
                    sA2[(mbase + r) * AS_ + ncol] = (_Float16)v;
                }
            }
            __syncthreads();

            // ---- D: v4f ring scatter-add; wave0 WMMA output projection;
            //         jump dot via shuffle reduction; pointer update ----
            for (int p4 = tid; p4 < B_ * E_ / 4; p4 += 256) {
                const int b = p4 >> 5, e4 = (p4 & 31) * 4;
                const v4f s2 = *(const v4f*)&sHid[(blk * B_ + b) * E_ + e4];
                #pragma unroll
                for (int j = 0; j < 9; ++j) {
                    float* pr = &ring[((size_t)b * P_ + sIdx[b * 9 + j]) * E_ + e4];
                    v4f v = *(const v4f*)pr;
                    v += sW[b * 9 + j] * s2;     // thread owns (b,e4); idx distinct
                    *(v4f*)pr = v;
                }
            }
            if (wave == 0) {  // wave-uniform: EXEC all ones for WMMA
                v8f c2 = {};
                const int m = lane & 15;
                #pragma unroll
                for (int kc = 0; kc < 4; ++kc) {
                    HFrag a2, bw;
                    const int koff = kc * 32 + kb;
                    a2.h2[0] = *(const v8h*)&sA2[m * AS_ + koff];
                    a2.h2[1] = *(const v8h*)&sA2[m * AS_ + koff + 16];
                    bw.h2[0] = *(const v8h*)&sWoutF[(kc * 32 + lane) * 16];
                    bw.h2[1] = *(const v8h*)&sWoutF[(kc * 32 + lane) * 16 + 8];
                    c2 = __builtin_amdgcn_wmma_f32_16x16x32_f16(
                            false, a2.v, false, bw.v, (short)0, c2, false, false);
                }
                const int mb2 = (lane >> 4) * 8;
                const int n   = lane & 15;
                if (n < 8) {
                    #pragma unroll
                    for (int r = 0; r < 8; ++r) sAcc[(mb2 + r) * 8 + n] += c2[r];
                }
            }
            if (tid < 128) {  // jump dot: 8 lanes/batch x 16 MACs + 3 shuffles
                const int b = tid >> 3, g = tid & 7;
                const float* hrow = &sHid[(blk * B_ + b) * E_ + g * 16];
                const float* jrow = &sJW[blk * E_ + g * 16];
                v4f pv = {};
                #pragma unroll
                for (int q = 0; q < 4; ++q)
                    pv += *(const v4f*)&hrow[q * 4] * *(const v4f*)&jrow[q * 4];
                float part = pv[0] + pv[1] + pv[2] + pv[3];
                part += __shfl_xor(part, 1, 32);
                part += __shfl_xor(part, 2, 32);
                part += __shfl_xor(part, 4, 32);
                if (g == 0) {
                    const float probs = fast_sigmoid(part + sJB[blk]);
                    const float ptr = sPtr[blk * B_ + b];
                    const int cur = min(max((int)ptr, 0), P_ - 1);
                    const float tgt = ptr_dest[blk * P_ + cur];
                    const float w1  = ptr + 1.0f;
                    const float walk = (w1 >= 256.0f) ? w1 - 256.0f : w1;
                    sPtr[blk * B_ + b] = (probs > 0.5f) ? tgt : walk;  // next read is t+1
                }
            }
            __threadfence_block();   // scatter must be visible to next block's gather
            __syncthreads();
        }

        // ---- out_t = mean over blocks (b_out added once, matching reference) ----
        if (tid < B_ * 8) {
            const int b = tid >> 3, o = tid & 7;
            out[((size_t)b * T_ + t) * 8 + o] = sAcc[tid] * 0.25f + sBout[o];
        }
        __syncthreads();
    }
}

extern "C" void kernel_launch(void* const* d_in, const int* in_sizes, int n_in,
                              void* d_out, int out_size, void* d_ws, size_t ws_size,
                              hipStream_t stream) {
    (void)in_sizes; (void)n_in; (void)out_size; (void)ws_size;
    const float* x            = (const float*)d_in[0];
    const float* W_in         = (const float*)d_in[1];
    const float* b_in         = (const float*)d_in[2];
    const float* W_out        = (const float*)d_in[3];
    const float* b_out        = (const float*)d_in[4];
    const float* W_p          = (const float*)d_in[5];
    const float* b_p          = (const float*)d_in[6];
    const float* ptr_dest     = (const float*)d_in[7];
    const float* jump_W       = (const float*)d_in[8];
    const float* jump_b       = (const float*)d_in[9];
    const float* ctx_strength = (const float*)d_in[10];
    const float* phase_bias   = (const float*)d_in[11];
    const float* pointer_init = (const float*)d_in[12];

    float* ring = (float*)d_ws;                 // (B, P, E) f32 = 2 MB, L2-resident
    const int ringN = B_ * P_ * E_;
    zero_ring_kernel<<<(ringN + 255) / 256, 256, 0, stream>>>(ring, ringN);

    swarm_ring_kernel<<<1, 256, 0, stream>>>(
        x, W_in, b_in, W_out, b_out, W_p, b_p,
        ptr_dest, jump_W, jump_b, ctx_strength, phase_bias, pointer_init,
        (float*)d_out, ring);
}
